// HFMelSpectrogram_25692494364951
// MI455X (gfx1250) — matrix operand order repaired and checked by reading, not compiled, not run on hardware
//
#include <hip/hip_runtime.h>
#include <hip/hip_bf16.h>

typedef __attribute__((ext_vector_type(16))) _Float16 v16h;
typedef __attribute__((ext_vector_type(8)))  _Float16 v8h;
typedef __attribute__((ext_vector_type(8)))  float    v8f;

#define N_FFT     1024
#define HOP       480
#define NBINS     513          // N_FFT/2 + 1
#define NB_FRAMES 1000
#define N_MELS    64
#define SPEC_W    1024
#define BATCH     32
#define WAVE_LEN  480000
#define PAD       512
#define BIN_PAD   544          // 513 rounded up to 34*16
#define FRAME_TILES 63         // ceil(1000/16)
#define BIN_TILES   33         // ceil(513/16)
#define WIN_SAMPS (15*HOP + N_FFT)   // 8224 samples cover 16 overlapping frames

// ---------------------------------------------------------------------------
// Kernel 1: f32 -> f16 weight conversion with zero padding to BIN_PAD rows
// per half (real rows 0..543, imag rows 544..1087). Row-major [row][k].
// ---------------------------------------------------------------------------
__global__ void __launch_bounds__(256)
pack_weights_f16(const float* __restrict__ w, _Float16* __restrict__ w16) {
    int idx = blockIdx.x * 256 + threadIdx.x;
    const int total = 2 * BIN_PAD * N_FFT;
    if (idx >= total) return;
    int k    = idx & (N_FFT - 1);
    int row  = idx >> 10;            // 0..1087
    int half = row / BIN_PAD;        // 0 = real, 1 = imag
    int bin  = row % BIN_PAD;
    float v = 0.0f;
    if (bin < NBINS) v = w[(size_t)(half * NBINS + bin) * N_FFT + k];
    w16[idx] = (_Float16)v;
}

// ---------------------------------------------------------------------------
// Kernel 2: STFT GEMM via v_wmma_f32_16x16x32_f16 + fused power = re^2+im^2.
// grid = (63 frame tiles, 5 bin-tile groups, 32 batches), block = 256 (8 waves).
// Wave w handles bin tile (blockIdx.y*8 + w). All waves share one LDS window
// of 8224 f16 samples (reflect-padded) covering the 16 frames of the tile.
// ---------------------------------------------------------------------------
__global__ void __launch_bounds__(256)
stft_power_wmma(const float* __restrict__ x, const _Float16* __restrict__ w16,
                float* __restrict__ power) {
    __shared__ _Float16 sA[WIN_SAMPS];

    const int ft = blockIdx.x;                 // frame tile
    const int b  = blockIdx.z;                 // batch
    const int t0 = ft * 16;
    const long start = (long)t0 * HOP - PAD;   // first padded-sample index
    const float* xb = x + (size_t)b * WAVE_LEN;

    // Cooperative staging: global f32 -> LDS f16, reflect-pad index math.
    for (int i = threadIdx.x; i < WIN_SAMPS; i += 256) {
        long j = start + i;
        if (j < 0) j = -j;
        if (j >= WAVE_LEN) j = 2L * WAVE_LEN - 2 - j;
        sA[i] = (_Float16)xb[j];
    }
    __syncthreads();

    const int wave = threadIdx.x >> 5;
    const int lane = threadIdx.x & 31;
    const int jt   = blockIdx.y * 8 + wave;    // bin tile
    if (jt >= BIN_TILES) return;

    const int hi  = lane >> 4;                 // lane half (0/1)
    const int lo  = lane & 15;                 // M (frame) for A; N (bin) for B/C
    const int bin = jt * 16 + lo;

    // B rows: real row `bin`, imag row `BIN_PAD + bin` (f16, row-major).
    const _Float16* wr = w16 + (size_t)bin * N_FFT;
    const _Float16* wi = w16 + (size_t)(BIN_PAD + bin) * N_FFT;

    // A fragment base: frame (t0 + lo) starts at window offset lo*HOP.
    const int aBase = lo * HOP;

    v8f cR = {};
    v8f cI = {};
    for (int k0 = 0; k0 < N_FFT; k0 += 32) {
        // A k order used here (contiguous, same as B): halves 0..7 = K(hi*16 .. hi*16+7),
        //                        halves 8..15 = K(hi*16+8 .. hi*16+15)
        const v8h a0 = *(const v8h*)&sA[aBase + k0 + hi * 16];
        const v8h a1 = *(const v8h*)&sA[aBase + k0 + hi * 16 + 8];
        v16h a = __builtin_shufflevector(a0, a1,
                                         0, 1, 2, 3, 4, 5, 6, 7,
                                         8, 9, 10, 11, 12, 13, 14, 15);
        // B k order used here (same as A): lanes 0-15 hold K=0..15, lanes 16-31 K=16..31,
        // contiguous along K per lane -> one 32B row-segment load.
        v16h br = *(const v16h*)&wr[k0 + hi * 16];
        v16h bi = *(const v16h*)&wi[k0 + hi * 16];

        cR = __builtin_amdgcn_wmma_f32_16x16x32_f16(false, a, false, br,
                                                    (short)0, cR, false, false);
        cI = __builtin_amdgcn_wmma_f32_16x16x32_f16(false, a, false, bi,
                                                    (short)0, cI, false, false);
    }

    if (bin >= NBINS) return;
    // C/D layout: VGPR r, this lane -> (frame t0 + hi*8 + r, bin). The 8
    // accumulator entries are consecutive frames -> contiguous along t.
    float* pb = power + ((size_t)b * NBINS + bin) * NB_FRAMES;
    #pragma unroll
    for (int r = 0; r < 8; ++r) {
        int t = t0 + hi * 8 + r;
        if (t < NB_FRAMES) pb[t] = cR[r] * cR[r] + cI[r] * cI[r];
    }
}

// ---------------------------------------------------------------------------
// Kernel 3: mel projection (K=513) + 10*log10. Coalesced over t; power slab
// is L2-resident (65 MB << 192 MB L2) across the 64 m-blocks that reuse it.
// lm layout: [b][m][t] (t contiguous).
// ---------------------------------------------------------------------------
__global__ void __launch_bounds__(128)
mel_log(const float* __restrict__ power, const float* __restrict__ melf,
        float* __restrict__ lm) {
    const int t = blockIdx.x * 128 + threadIdx.x;
    const int m = blockIdx.y;
    const int b = blockIdx.z;
    if (t >= NB_FRAMES) return;
    const float* p = power + (size_t)b * NBINS * NB_FRAMES + t;
    float acc = 0.0f;
    for (int f = 0; f < NBINS; ++f)
        acc = fmaf(p[(size_t)f * NB_FRAMES], melf[f * N_MELS + m], acc);
    lm[((size_t)b * N_MELS + m) * NB_FRAMES + t] =
        10.0f * log10f(fmaxf(acc, 1e-10f));
}

// ---------------------------------------------------------------------------
// Kernel 4: bilinear align_corners resize 1000 -> 1024 rows; width 64 -> 64 is
// the identity, so only vertical lerp. Also performs the (b,m,t)->(b,h,m)
// transpose into the output layout (32,1,1024,64).
// ---------------------------------------------------------------------------
__global__ void __launch_bounds__(256)
resize_bilinear(const float* __restrict__ lm, float* __restrict__ out) {
    int idx = blockIdx.x * 256 + threadIdx.x;
    if (idx >= BATCH * SPEC_W * N_MELS) return;
    int m = idx & 63;
    int h = (idx >> 6) & (SPEC_W - 1);
    int b = idx >> 16;
    float scale = (float)(NB_FRAMES - 1) / (float)(SPEC_W - 1);
    float pos = (float)h * scale;
    int h0 = (int)floorf(pos);
    h0 = min(max(h0, 0), NB_FRAMES - 1);
    float hw = pos - (float)h0;
    int h1 = min(h0 + 1, NB_FRAMES - 1);
    const float* base = lm + ((size_t)b * N_MELS + m) * NB_FRAMES;
    out[idx] = base[h0] * (1.0f - hw) + base[h1] * hw;
}

// ---------------------------------------------------------------------------
// Workspace layout (bytes):
//   [0, 2228224)               f16 packed weights   2*544*1024*2
//   [2228224, 67892224)        power  f32 [32][513][1000]
//   [67892224, 76084224)       logmel f32 [32][64][1000]
// ---------------------------------------------------------------------------
extern "C" void kernel_launch(void* const* d_in, const int* in_sizes, int n_in,
                              void* d_out, int out_size, void* d_ws, size_t ws_size,
                              hipStream_t stream) {
    const float* wav   = (const float*)d_in[0];   // (32, 480000) f32
    const float* stftw = (const float*)d_in[1];   // (1026, 1024) f32
    const float* melf  = (const float*)d_in[2];   // (513, 64)    f32
    float* out = (float*)d_out;                   // (32,1,1024,64) f32

    char* ws = (char*)d_ws;
    _Float16* w16  = (_Float16*)ws;
    float*   power = (float*)(ws + 2228224);
    float*   lm    = (float*)(ws + 67892224);

    {
        int total = 2 * BIN_PAD * N_FFT;
        pack_weights_f16<<<dim3((total + 255) / 256), 256, 0, stream>>>(stftw, w16);
    }
    stft_power_wmma<<<dim3(FRAME_TILES, (BIN_TILES + 7) / 8, BATCH), 256, 0, stream>>>(
        wav, w16, power);
    mel_log<<<dim3((NB_FRAMES + 127) / 128, N_MELS, BATCH), 128, 0, stream>>>(
        power, melf, lm);
    {
        int total = BATCH * SPEC_W * N_MELS;
        resize_bilinear<<<dim3(total / 256), 256, 0, stream>>>(lm, out);
    }
}
